// KHeadAttnCosSimilarity_1262720385499
// MI455X (gfx1250) — matrix operand order repaired and checked by reading, not compiled, not run on hardware
//
#include <hip/hip_runtime.h>

#define HEAD_NUM 12
#define D_KK 64
#define LSEQ 512
#define DMODEL 768
#define BATCH 8
#define MROWS (BATCH * LSEQ)   // 4096

typedef __attribute__((ext_vector_type(16))) __bf16 v16bf;
typedef __attribute__((ext_vector_type(8)))  float  v8f;

union FragBF { v16bf v; unsigned short u[16]; };

__device__ __forceinline__ unsigned short f2bf(float f) {
    unsigned int u = __float_as_uint(f);
    u += 0x7FFFu + ((u >> 16) & 1u);   // round-to-nearest-even
    return (unsigned short)(u >> 16);
}

__device__ __forceinline__ uint2 f2bf4(float4 v) {
    uint2 r;
    r.x = (unsigned)f2bf(v.x) | ((unsigned)f2bf(v.y) << 16);
    r.y = (unsigned)f2bf(v.z) | ((unsigned)f2bf(v.w) << 16);
    return r;
}

// A-matrix fragment (16x32 bf16, wave32):
// lanes 0-15: row M=lane, K={0..7} in elems 0..7, {16..23} in elems 8..15
// lanes 16-31: row M=lane-16, K={8..15} and {24..31}
__device__ __forceinline__ v16bf load_a_frag(const unsigned short* base, int stride,
                                             int row0, int lane, int kbase) {
    const unsigned short* rp = base + (row0 + (lane & 15)) * stride + kbase + ((lane >> 4) << 3);
    FragBF f;
#pragma unroll
    for (int e = 0; e < 8; ++e) f.u[e] = rp[e];
#pragma unroll
    for (int e = 0; e < 8; ++e) f.u[8 + e] = rp[16 + e];
    return f.v;
}

// B-matrix fragment (32x16 bf16, wave32), staged as B^T rows (n-major, k contiguous):
// lanes 0-15: col N=lane, K={0..7} in elems 0..7, {16..23} in elems 8..15; lanes 16-31: col N=lane-16, K={8..15} and {24..31}
__device__ __forceinline__ v16bf load_b_frag(const unsigned short* base, int stride,
                                             int col0, int lane, int kbase) {
    const unsigned short* rp = base + (col0 + (lane & 15)) * stride + kbase + ((lane >> 4) << 3);
    FragBF f;
#pragma unroll
    for (int e = 0; e < 8; ++e) f.u[e] = rp[e];
#pragma unroll
    for (int e = 0; e < 8; ++e) f.u[8 + e] = rp[16 + e];
    return f.v;
}

// C = A(4096x768) @ W^T(768x768) + bias, f32 in memory, bf16 WMMA math.
// block = 256 threads (8 waves as 2x4); block tile 64x64, each wave owns two
// 16-row A fragments sharing one B fragment -> 2 WMMAs per barrier pair.
__global__ __launch_bounds__(256)
void linear_bf16_wmma(const float* __restrict__ A, const float* __restrict__ W,
                      const float* __restrict__ bias, float* __restrict__ C) {
    __shared__ unsigned short aT[64 * 40];  // [m][k] padded
    __shared__ unsigned short wT[64 * 40];  // [n][k] padded  (== B^T)
    const int tid  = threadIdx.x;
    const int lane = tid & 31;
    const int wave = tid >> 5;
    const int wm   = wave & 1;
    const int wn   = wave >> 1;
    const int m0   = blockIdx.x * 64;
    const int n0   = blockIdx.y * 64;

    v8f acc0 = {};
    v8f acc1 = {};
    for (int k0 = 0; k0 < DMODEL; k0 += 32) {
        float4 fa[2], fw[2];
#pragma unroll
        for (int u = 0; u < 2; ++u) {                 // 64x32 = 2048 elems = 256*8
            int idx = tid * 4 + u * 1024;
            int r = idx >> 5, c = idx & 31;
            fa[u] = *(const float4*)(A + (long)(m0 + r) * DMODEL + k0 + c);
            fw[u] = *(const float4*)(W + (long)(n0 + r) * DMODEL + k0 + c);
        }
#pragma unroll
        for (int u = 0; u < 2; ++u) {
            int idx = tid * 4 + u * 1024;
            int r = idx >> 5, c = idx & 31;
            *(uint2*)(aT + r * 40 + c) = f2bf4(fa[u]);
            *(uint2*)(wT + r * 40 + c) = f2bf4(fw[u]);
        }
        __syncthreads();
        v16bf a0 = load_a_frag(aT, 40, wm * 16, lane, 0);
        v16bf a1 = load_a_frag(aT, 40, wm * 16 + 32, lane, 0);
        v16bf b  = load_b_frag(wT, 40, wn * 16, lane, 0);
        acc0 = __builtin_amdgcn_wmma_f32_16x16x32_bf16(false, a0, false, b, (short)0, acc0, false, false);
        acc1 = __builtin_amdgcn_wmma_f32_16x16x32_bf16(false, a1, false, b, (short)0, acc1, false, false);
        __syncthreads();
    }

    const int   nl = n0 + wn * 16 + (lane & 15);
    const float bv = bias[nl];
    float* cp = C + (long)(m0 + wm * 16 + ((lane >> 4) << 3)) * DMODEL + nl;
#pragma unroll
    for (int r = 0; r < 8; ++r) cp[r * DMODEL] = acc0[r] + bv;        // imm offsets
    cp += 32 * DMODEL;
#pragma unroll
    for (int r = 0; r < 8; ++r) cp[r * DMODEL] = acc1[r] + bv;
}

// Per (b,h): dots = sem.sem^T, scores = q.k^T/8, mask, cosine threshold,
// row softmax, out = cos_adj * p_attn * row_keep.
// grid = (L/32, B*H), block = 256 (8 waves as 2x4 over a 32x64 tile per j-step).
__global__ __launch_bounds__(256)
void attn_cos_wmma(const float* __restrict__ mapped, const float* __restrict__ qb,
                   const float* __restrict__ kb, const unsigned char* __restrict__ mask,
                   float* __restrict__ out) {
    extern __shared__ char smem_raw[];
    float*          norms  = (float*)smem_raw;                                  // 2048 B
    unsigned short* semI   = (unsigned short*)(smem_raw + 2048);                // 32*72*2
    unsigned short* qI     = (unsigned short*)(smem_raw + 2048 + 4608);         // 32*72*2
    unsigned short* semJ   = (unsigned short*)(smem_raw + 2048 + 2 * 4608);     // 64*72*2
    unsigned short* kJ     = (unsigned short*)(smem_raw + 2048 + 2 * 4608 + 9216);
    float*          scores = (float*)(smem_raw + 2048 + 2 * 4608 + 2 * 9216);   // 32*520*4
    unsigned char*  cosAdj = (unsigned char*)(smem_raw + 2048 + 2 * 4608 + 2 * 9216 + 66560);

    const int tid  = threadIdx.x;
    const int lane = tid & 31;
    const int wave = tid >> 5;
    const int bh   = blockIdx.y;            // b*H + h
    const int b    = bh / HEAD_NUM;
    const int h    = bh % HEAD_NUM;
    const int i0   = blockIdx.x * 32;

    const long rowBase = (long)b * LSEQ;
    const float* semP = mapped + rowBase * DMODEL + h * D_KK;
    const float* qP   = qb     + rowBase * DMODEL + h * D_KK;
    const float* kP   = kb     + rowBase * DMODEL + h * D_KK;
    const unsigned char* maskP = mask + (long)b * LSEQ * LSEQ;

    // per-(b,h) row norms of sem (vectorized, 2 rows/thread)
#pragma unroll
    for (int u = 0; u < 2; ++u) {
        int j = tid + u * 256;
        const float4* p = (const float4*)(semP + (long)j * DMODEL);
        float s = 0.f;
#pragma unroll
        for (int c = 0; c < 16; ++c) {
            float4 v = p[c];
            s += v.x * v.x + v.y * v.y + v.z * v.z + v.w * v.w;
        }
        norms[j] = sqrtf(s);
    }

    // stage the 32 i-rows of sem and q (bf16); 2048 elems each = 256*8
    {
        float4 fs[2], fq[2];
#pragma unroll
        for (int u = 0; u < 2; ++u) {
            int idx = tid * 4 + u * 1024;
            int r = idx >> 6, c = idx & 63;
            fs[u] = *(const float4*)(semP + (long)(i0 + r) * DMODEL + c);
            fq[u] = *(const float4*)(qP   + (long)(i0 + r) * DMODEL + c);
        }
#pragma unroll
        for (int u = 0; u < 2; ++u) {
            int idx = tid * 4 + u * 1024;
            int r = idx >> 6, c = idx & 63;
            *(uint2*)(semI + r * 72 + c) = f2bf4(fs[u]);
            *(uint2*)(qI   + r * 72 + c) = f2bf4(fq[u]);
        }
    }
    __syncthreads();

    const int wm = wave & 1;
    const int wn = wave >> 1;

    for (int jt = 0; jt < LSEQ / 64; ++jt) {
        const int j0 = jt * 64;
        {   // stage 64 j-rows of sem and k: 4096 elems each = 256*16
            float4 fs[4], fk[4];
#pragma unroll
            for (int u = 0; u < 4; ++u) {
                int idx = tid * 4 + u * 1024;
                int r = idx >> 6, c = idx & 63;
                fs[u] = *(const float4*)(semP + (long)(j0 + r) * DMODEL + c);
                fk[u] = *(const float4*)(kP   + (long)(j0 + r) * DMODEL + c);
            }
#pragma unroll
            for (int u = 0; u < 4; ++u) {
                int idx = tid * 4 + u * 1024;
                int r = idx >> 6, c = idx & 63;
                *(uint2*)(semJ + r * 72 + c) = f2bf4(fs[u]);
                *(uint2*)(kJ   + r * 72 + c) = f2bf4(fk[u]);
            }
        }
        if (jt + 1 < LSEQ / 64) {   // global_prefetch_b8 for the next j-tile
            __builtin_prefetch(semP + (long)(j0 + 64 + (tid & 63)) * DMODEL, 0, 0);
            __builtin_prefetch(kP   + (long)(j0 + 64 + (tid & 63)) * DMODEL, 0, 0);
        }
        __syncthreads();

        v8f accD = {};
        v8f accS = {};
#pragma unroll
        for (int kk = 0; kk < D_KK; kk += 32) {
            v16bf aS = load_a_frag(semI, 72, wm * 16, lane, kk);
            v16bf bS = load_b_frag(semJ, 72, wn * 16, lane, kk);
            accD = __builtin_amdgcn_wmma_f32_16x16x32_bf16(false, aS, false, bS, (short)0, accD, false, false);
            v16bf aQ = load_a_frag(qI, 72, wm * 16, lane, kk);
            v16bf bK = load_b_frag(kJ, 72, wn * 16, lane, kk);
            accS = __builtin_amdgcn_wmma_f32_16x16x32_bf16(false, aQ, false, bK, (short)0, accS, false, false);
        }

        const int j = j0 + wn * 16 + (lane & 15);
#pragma unroll
        for (int r = 0; r < 8; ++r) {
            int m = wm * 16 + ((lane >> 4) << 3) + r;
            int i = i0 + m;
            unsigned char mk = maskP[(long)i * LSEQ + j];
            float denom = fmaxf(norms[i] * norms[j], 1e-8f);
            float cosv  = mk ? 0.f : (accD[r] / denom);
            cosAdj[m * LSEQ + j] = (cosv > 0.3f) ? 1 : 0;
            scores[m * 520 + j]  = mk ? -1e10f : (accS[r] * 0.125f);  // 1/sqrt(64)
        }
        __syncthreads();
    }

    // softmax over the full 512-wide row; 8 threads/row, each owns a contiguous
    // 64-float chunk (b128 LDS traffic), shuffle reductions across the 8.
    const int row   = tid >> 3;
    const int s8    = tid & 7;
    const int i     = i0 + row;
    float* srow     = scores + row * 520 + s8 * 64;
    float mx = -3.4e38f;
#pragma unroll
    for (int t = 0; t < 64; t += 4) {
        float4 v = *(const float4*)(srow + t);
        mx = fmaxf(mx, fmaxf(fmaxf(v.x, v.y), fmaxf(v.z, v.w)));
    }
#pragma unroll
    for (int d = 4; d > 0; d >>= 1) mx = fmaxf(mx, __shfl_xor(mx, d, 8));
    float sum = 0.f;
#pragma unroll
    for (int t = 0; t < 64; t += 4) {
        float4 v = *(const float4*)(srow + t);
        v.x = __expf(v.x - mx); v.y = __expf(v.y - mx);
        v.z = __expf(v.z - mx); v.w = __expf(v.w - mx);
        *(float4*)(srow + t) = v;
        sum += v.x + v.y + v.z + v.w;
    }
#pragma unroll
    for (int d = 4; d > 0; d >>= 1) sum += __shfl_xor(sum, d, 8);
    const float inv = 1.f / sum;
    const float rk  = maskP[(long)i * LSEQ] ? 0.f : 1.f;   // row_keep = !mask[b,i,0]
    const unsigned char* crow = cosAdj + row * LSEQ + s8 * 64;
    float* orow = out + ((long)bh * LSEQ + i) * LSEQ + s8 * 64;
#pragma unroll
    for (int t = 0; t < 64; t += 4) {
        float4 v = *(const float4*)(srow + t);
        uchar4 cm = *(const uchar4*)(crow + t);
        float4 res;
        res.x = cm.x ? (v.x * inv * rk) : 0.f;
        res.y = cm.y ? (v.y * inv * rk) : 0.f;
        res.z = cm.z ? (v.z * inv * rk) : 0.f;
        res.w = cm.w ? (v.w * inv * rk) : 0.f;
        *(float4*)(orow + t) = res;                        // global_store_b128
    }
}

extern "C" void kernel_launch(void* const* d_in, const int* in_sizes, int n_in,
                              void* d_out, int out_size, void* d_ws, size_t ws_size,
                              hipStream_t stream) {
    (void)in_sizes; (void)n_in; (void)out_size; (void)ws_size;
    const float* embs          = (const float*)d_in[0];
    const unsigned char* mask  = (const unsigned char*)d_in[1];  // bool array (1 byte/elem)
    const float* W_map         = (const float*)d_in[2];
    const float* b_map         = (const float*)d_in[3];
    const float* W_q           = (const float*)d_in[4];
    const float* b_q           = (const float*)d_in[5];
    const float* W_k           = (const float*)d_in[6];
    const float* b_k           = (const float*)d_in[7];
    float* outp                = (float*)d_out;

    float* mapped = (float*)d_ws;
    float* qbuf   = mapped + (size_t)MROWS * DMODEL;
    float* kbuf   = qbuf   + (size_t)MROWS * DMODEL;   // total ws use: ~37.8 MB

    dim3 blk(256);
    dim3 ggrid(MROWS / 64, DMODEL / 64);
    linear_bf16_wmma<<<ggrid, blk, 0, stream>>>(embs,   W_map, b_map, mapped);
    linear_bf16_wmma<<<ggrid, blk, 0, stream>>>(mapped, W_q,   b_q,   qbuf);
    linear_bf16_wmma<<<ggrid, blk, 0, stream>>>(mapped, W_k,   b_k,   kbuf);

    const size_t smem = 2048 + 2 * 4608 + 2 * 9216 + 66560 + 16384;  // 112640 B (<320KB WGP LDS)
    dim3 agrid(LSEQ / 32, BATCH * HEAD_NUM);
    attn_cos_wmma<<<agrid, blk, smem, stream>>>(mapped, qbuf, kbuf, mask, outp);
}
